// ActorGNN_69922067578970
// MI455X (gfx1250) — compile-verified, matrix operand order checked by reading, not run on hardware
//
#include <hip/hip_runtime.h>
#include <hip/hip_bf16.h>

// ActorGNN on MI455X (gfx1250, wave32, WMMA).
//
// The whole working set (xh 25.6MB f16, agg 51.2MB f32, 72KB repacked weights)
// is L2-resident (192MB), so the edge stage is issue/L2-bound, not HBM-bound.
// Edge kernel therefore processes TWO 16-edge M-tiles per wave so every
// B fragment load feeds two independent v_wmma_f32_16x16x32_f16 ops:
// halves weight-load issue per edge and breaks accumulator RAW chains.

typedef __attribute__((ext_vector_type(16))) _Float16 v16h;
typedef __attribute__((ext_vector_type(8)))  float    v8f;

union VFrag { v16h v; uint4 q[2]; };

#define ROWP_E 304   // 288 halves + pad (608B rows, 16B aligned)
#define ROWP_N 272   // 256 halves + pad
#define EPB    128   // edges per block (4 waves x 32 edges)

// ---------------- prep kernels ----------------

__global__ void k_count(const int* __restrict__ dst, unsigned* __restrict__ cnt, int E) {
    int e = blockIdx.x * blockDim.x + threadIdx.x;
    if (e < E) atomicAdd(&cnt[dst[e]], 1u);
}

__global__ void k_f32_to_f16(const float* __restrict__ in, _Float16* __restrict__ out, long n) {
    long i = (long)blockIdx.x * blockDim.x + threadIdx.x;
    if (i < n) out[i] = (_Float16)in[i];
}

// Repack W (K x 128 row-major f32) into fragment-linear f16:
// out[((kt*8+nt)*32+lane)*16 + j] = W[k*128 + n], zero-padded past K.
// 16-bit operand layout: k_local = j + (j>=8?8:0) + (lane>=16?8:0).
__global__ void k_repack(const float* __restrict__ W, _Float16* __restrict__ out,
                         int K, int total) {
    int i = blockIdx.x * blockDim.x + threadIdx.x;
    if (i >= total) return;
    int j    = i & 15;
    int lane = (i >> 4) & 31;
    int nt   = (i >> 9) & 7;
    int kt   = i >> 12;
    int kl = j + ((j >= 8) ? 8 : 0) + ((lane >= 16) ? 8 : 0);
    int k  = kt * 32 + kl;
    int n  = nt * 16 + (lane & 15);
    float v = (k < K) ? W[k * 128 + n] : 0.0f;
    out[i] = (_Float16)v;
}

// ---------------- edge message kernel (WMMA) ----------------
// 128 threads = 4 waves; 128 edges per block; 32 edges (two M-tiles) per wave.
// Dynamic LDS: 128*ROWP_E halves + 128 dst ints = 78,336 bytes.

__global__ void __launch_bounds__(128)
k_edge_wmma(const _Float16* __restrict__ xh,
            const int* __restrict__ src, const int* __restrict__ dst,
            const float* __restrict__ ea,
            const _Float16* __restrict__ WeF,   // 9 KT x 8 NT fragments
            const float* __restrict__ be,
            float* __restrict__ agg, int E) {
    extern __shared__ char smem[];
    _Float16* sA   = (_Float16*)smem;
    int*      sDst = (int*)(smem + (size_t)EPB * ROWP_E * 2);

    const int  t    = threadIdx.x;
    const long base = (long)blockIdx.x * EPB;

    // ---- gather: one thread per edge row ----
    {
        const long eg = base + t;
        if (eg < E) {
            const int s = src[eg];
            const int d = dst[eg];
            sDst[t] = d;
            const uint4* sp = (const uint4*)(xh + (long)s * 128);
            const uint4* dp = (const uint4*)(xh + (long)d * 128);
            uint4* rowq = (uint4*)(sA + t * ROWP_E);
            #pragma unroll
            for (int j = 0; j < 16; ++j) rowq[j] = sp[j];        // halves [0,128)
            #pragma unroll
            for (int j = 0; j < 16; ++j) rowq[16 + j] = dp[j];   // halves [128,256)
            const float* ep = ea + eg * 16;
            _Float16* row = sA + t * ROWP_E;
            #pragma unroll
            for (int j = 0; j < 16; ++j) row[256 + j] = (_Float16)ep[j];
            #pragma unroll
            for (int j = 0; j < 16; ++j) row[272 + j] = (_Float16)0.0f; // zero K-pad
        }
    }
    __syncthreads();

    // ---- compute: two 16x288 @ 288x128 per wave ----
    const int lane = t & 31;
    const int wv   = t >> 5;
    if (base + (wv + 1) * 32 <= (long)E) {  // wave-uniform; EXEC all-ones inside
        const int hi   = lane >> 4;
        const int col0 = lane & 15;
        const _Float16* Arow0 = sA + (wv * 32 + col0) * ROWP_E;
        const _Float16* Arow1 = Arow0 + 16 * ROWP_E;

        VFrag af0[9], af1[9];
        #pragma unroll
        for (int kt = 0; kt < 9; ++kt) {
            const uint4* p0 = (const uint4*)(Arow0 + kt * 32 + hi * 8);
            af0[kt].q[0] = p0[0];   // halves [k0+hi*8, +8)
            af0[kt].q[1] = p0[2];   // halves [k0+16+hi*8, +8)
            const uint4* p1 = (const uint4*)(Arow1 + kt * 32 + hi * 8);
            af1[kt].q[0] = p1[0];
            af1[kt].q[1] = p1[2];
        }
        const v16h* B = (const v16h*)WeF;

        #pragma unroll
        for (int nt = 0; nt < 8; ++nt) {
            const float bias = be[nt * 16 + col0];
            v8f acc0, acc1;
            #pragma unroll
            for (int r = 0; r < 8; ++r) { acc0[r] = bias; acc1[r] = bias; }
            #pragma unroll
            for (int kt = 0; kt < 9; ++kt) {
                v16h b = B[(kt * 8 + nt) * 32 + lane];
                acc0 = __builtin_amdgcn_wmma_f32_16x16x32_f16(
                         false, af0[kt].v, false, b, (short)0, acc0, false, false);
                acc1 = __builtin_amdgcn_wmma_f32_16x16x32_f16(
                         false, af1[kt].v, false, b, (short)0, acc1, false, false);
            }
            const int c = nt * 16 + col0;
            #pragma unroll
            for (int r = 0; r < 8; ++r) {
                const int m0 = wv * 32 + r + hi * 8;         // C layout row, tile 0
                float v0 = fmaxf(acc0[r], 0.0f);
                atomicAdd(agg + (long)sDst[m0] * 128 + c, v0);
                float v1 = fmaxf(acc1[r], 0.0f);
                atomicAdd(agg + (long)sDst[m0 + 16] * 128 + c, v1);
            }
        }
    }
}

// ---------------- node update kernel (WMMA) ----------------

__global__ void __launch_bounds__(128)
k_node_wmma(const _Float16* __restrict__ xh,
            const float* __restrict__ agg, const unsigned* __restrict__ cnt,
            const _Float16* __restrict__ WnF,   // 8 KT x 8 NT fragments
            const float* __restrict__ bn,
            _Float16* __restrict__ xout, int N) {
    __shared__ _Float16 sA[64 * ROWP_N];

    const int  t    = threadIdx.x;
    const long base = (long)blockIdx.x * 64;

    {
        const int  nl = t >> 1, part = t & 1;
        const long ng = base + nl;
        if (ng < N) {
            if (part == 0) {
                const uint4* sp = (const uint4*)(xh + ng * 128);
                uint4* dp = (uint4*)(sA + nl * ROWP_N);
                #pragma unroll
                for (int j = 0; j < 16; ++j) dp[j] = sp[j];           // halves [0,128)
            } else {
                const float rinv = 1.0f / fmaxf((float)cnt[ng], 1.0f);
                const float* ap = agg + ng * 128;
                _Float16* row = sA + nl * ROWP_N + 128;
                #pragma unroll
                for (int j = 0; j < 128; ++j) row[j] = (_Float16)(ap[j] * rinv);
            }
        }
    }
    __syncthreads();

    const int lane = t & 31;
    const int wv   = t >> 5;
    if (base + wv * 16 < (long)N) {        // N % 16 == 0; wave-uniform guard
        const int hi   = lane >> 4;
        const int col0 = lane & 15;
        const _Float16* Arow = sA + (wv * 16 + col0) * ROWP_N;

        VFrag af[8];
        #pragma unroll
        for (int kt = 0; kt < 8; ++kt) {
            const uint4* p = (const uint4*)(Arow + kt * 32 + hi * 8);
            af[kt].q[0] = p[0];
            af[kt].q[1] = p[2];
        }
        const v16h* B = (const v16h*)WnF;

        #pragma unroll
        for (int nt = 0; nt < 8; ++nt) {
            const float bias = bn[nt * 16 + col0];
            v8f acc;
            #pragma unroll
            for (int r = 0; r < 8; ++r) acc[r] = bias;
            #pragma unroll
            for (int kt = 0; kt < 8; ++kt) {
                v16h b = B[(kt * 8 + nt) * 32 + lane];
                acc = __builtin_amdgcn_wmma_f32_16x16x32_f16(
                        false, af[kt].v, false, b, (short)0, acc, false, false);
            }
            const int c = nt * 16 + col0;
            #pragma unroll
            for (int r = 0; r < 8; ++r) {
                float v = fmaxf(acc[r], 0.0f);            // act=True for blocks 1..3
                const long node = base + wv * 16 + r + hi * 8;
                xout[node * 128 + c] = (_Float16)v;
            }
        }
    }
}

// ---------------- block 4 (dout = 1): scalar ----------------

__global__ void k_edge4(const _Float16* __restrict__ xh,
                        const int* __restrict__ src, const int* __restrict__ dst,
                        const float* __restrict__ ea,
                        const float* __restrict__ We4, const float* __restrict__ be4,
                        float* __restrict__ agg4, int E) {
    int e = blockIdx.x * blockDim.x + threadIdx.x;
    if (e >= E) return;
    float s = be4[0];
    const _Float16* xs = xh + (long)src[e] * 128;
    const _Float16* xd = xh + (long)dst[e] * 128;
    #pragma unroll 8
    for (int k = 0; k < 128; ++k) s += (float)xs[k] * We4[k];
    #pragma unroll 8
    for (int k = 0; k < 128; ++k) s += (float)xd[k] * We4[128 + k];
    const float* ep = ea + (long)e * 16;
    #pragma unroll
    for (int j = 0; j < 16; ++j) s += ep[j] * We4[256 + j];
    s = fmaxf(s, 0.0f);
    atomicAdd(agg4 + dst[e], s);
}

__global__ void k_node4(const _Float16* __restrict__ xh,
                        const float* __restrict__ agg4, const unsigned* __restrict__ cnt,
                        const float* __restrict__ Wn4, const float* __restrict__ bn4,
                        float* __restrict__ out, int N) {
    int n = blockIdx.x * blockDim.x + threadIdx.x;
    if (n >= N) return;
    const float a = agg4[n] / fmaxf((float)cnt[n], 1.0f);
    float s = bn4[0] + a * Wn4[128];
    const _Float16* xr = xh + (long)n * 128;
    #pragma unroll 8
    for (int k = 0; k < 128; ++k) s += (float)xr[k] * Wn4[k];
    out[n] = 1.0f / (1.0f + __expf(-s));
}

// ---------------- host orchestration ----------------

extern "C" void kernel_launch(void* const* d_in, const int* in_sizes, int n_in,
                              void* d_out, int out_size, void* d_ws, size_t ws_size,
                              hipStream_t stream) {
    const float* x  = (const float*)d_in[0];
    const int*   ei = (const int*)d_in[1];
    const float* ea = (const float*)d_in[2];
    const float* We[4] = {(const float*)d_in[3],  (const float*)d_in[7],
                          (const float*)d_in[11], (const float*)d_in[15]};
    const float* be[4] = {(const float*)d_in[4],  (const float*)d_in[8],
                          (const float*)d_in[12], (const float*)d_in[16]};
    const float* Wn[4] = {(const float*)d_in[5],  (const float*)d_in[9],
                          (const float*)d_in[13], (const float*)d_in[17]};
    const float* bn[4] = {(const float*)d_in[6],  (const float*)d_in[10],
                          (const float*)d_in[14], (const float*)d_in[18]};

    const int N = in_sizes[0] / 128;
    const int E = in_sizes[2] / 16;
    const int* srcv = ei;
    const int* dstv = ei + E;

    // workspace carve-out (~104 MB)
    char* w = (char*)d_ws;
    size_t off = 0;
    auto alloc = [&](size_t bytes) -> void* {
        size_t o = (off + 255) & ~(size_t)255;
        off = o + bytes;
        return (void*)(w + o);
    };
    _Float16* xhA  = (_Float16*)alloc((size_t)N * 128 * 2);
    _Float16* xhB  = (_Float16*)alloc((size_t)N * 128 * 2);
    float*    agg  = (float*)   alloc((size_t)N * 128 * 4);
    unsigned* cnt  = (unsigned*)alloc((size_t)N * 4);
    float*    agg4 = (float*)   alloc((size_t)N * 4);
    _Float16* WeF[3]; _Float16* WnF[3];
    const int WE_TOT = 9 * 8 * 32 * 16;   // 36864 halves
    const int WN_TOT = 8 * 8 * 32 * 16;   // 32768 halves
    for (int l = 0; l < 3; ++l) WeF[l] = (_Float16*)alloc((size_t)WE_TOT * 2);
    for (int l = 0; l < 3; ++l) WnF[l] = (_Float16*)alloc((size_t)WN_TOT * 2);

    // degree (once) + f16 activations + weight repack
    hipMemsetAsync(cnt, 0, (size_t)N * 4, stream);
    k_count<<<(E + 255) / 256, 256, 0, stream>>>(dstv, cnt, E);
    {
        long nx = (long)N * 128;
        k_f32_to_f16<<<(unsigned)((nx + 255) / 256), 256, 0, stream>>>(x, xhA, nx);
    }
    for (int l = 0; l < 3; ++l) {
        k_repack<<<(WE_TOT + 255) / 256, 256, 0, stream>>>(We[l], WeF[l], 272, WE_TOT);
        k_repack<<<(WN_TOT + 255) / 256, 256, 0, stream>>>(Wn[l], WnF[l], 256, WN_TOT);
    }

    const size_t edgeLds = (size_t)EPB * ROWP_E * 2 + EPB * sizeof(int); // 78,336 B

    _Float16* cur = xhA;
    _Float16* nxt = xhB;
    for (int l = 0; l < 3; ++l) {
        hipMemsetAsync(agg, 0, (size_t)N * 128 * 4, stream);
        k_edge_wmma<<<(E + EPB - 1) / EPB, 128, edgeLds, stream>>>(
            cur, srcv, dstv, ea, WeF[l], be[l], agg, E);
        k_node_wmma<<<(N + 63) / 64, 128, 0, stream>>>(cur, agg, cnt, WnF[l], bn[l], nxt, N);
        _Float16* tmp = cur; cur = nxt; nxt = tmp;
    }

    hipMemsetAsync(agg4, 0, (size_t)N * 4, stream);
    k_edge4<<<(E + 255) / 256, 256, 0, stream>>>(cur, srcv, dstv, ea, We[3], be[3], agg4, E);
    k_node4<<<(N + 255) / 256, 256, 0, stream>>>(cur, agg4, cnt, Wn[3], bn[3],
                                                 (float*)d_out, N);
}
